// Pairwise_77927886618998
// MI455X (gfx1250) — matrix operand order repaired and checked by reading, not compiled, not run on hardware
//
#include <hip/hip_runtime.h>

// Problem constants (from the reference)
#define NE   1024   // entities
#define NIN  64     // input features
#define HID  128    // hidden
#define NOUT 64     // output
#define NAB  256    // AB matrix columns = 2*HID

typedef __attribute__((ext_vector_type(16))) _Float16 v16h;
typedef __attribute__((ext_vector_type(8)))  float    v8f;

__device__ __forceinline__ float fast_tanh(float x) {
#if __has_builtin(__builtin_amdgcn_tanhf)
  return __builtin_amdgcn_tanhf(x);          // v_tanh_f32 (gfx1250 transcendental)
#else
  // tanh(x) = 1 - 2/(exp(2x)+1), exp via v_exp_f32 (exp2), fast v_rcp_f32
  float e = __builtin_amdgcn_exp2f(x * 2.8853900817779268f); // 2*log2(e)*x
  return 1.0f - 2.0f * __builtin_amdgcn_rcpf(e + 1.0f);
#endif
}

// ---------------------------------------------------------------------------
// Kernel 1: AB[1024 x 256] = x[1024 x 64] @ W'[64 x 256]
//   W'[k][n] = (n < 128) ? W1[k][n] : W1[k+64][n-128]
//   => AB[:, :128] = x @ W1[:64]  (self part A)
//      AB[:, 128:] = x @ W1[64:]  (partner part B)
// One wave per 16x16 tile, v_wmma_f32_16x16x32_f16, K = 64 -> 2 steps.
// Grid: 1024 tiles / 4 waves = 256 blocks x 128 threads.
// ---------------------------------------------------------------------------
__global__ __launch_bounds__(128) void k1_gemm_xw1(
    const float* __restrict__ x, const float* __restrict__ W1,
    float* __restrict__ AB) {
  const int lane = threadIdx.x & 31;
  const int wave = blockIdx.x * (blockDim.x >> 5) + (threadIdx.x >> 5);
  const int tm = wave >> 4;          // 0..63  (row tile)
  const int tn = wave & 15;          // 0..15  (col tile over 256 cols)
  const int g  = lane >> 4;          // half-wave group
  const int l  = lane & 15;
  const int row = tm * 16 + l;       // A-frag row
  const int n   = tn * 16 + l;       // B-frag / output column (0..255)
  const int wcol = (n < HID) ? n : (n - HID);
  const int wrow_off = (n < HID) ? 0 : NIN;    // uniform per wave

  v8f c = {};
#pragma unroll
  for (int k0 = 0; k0 < NIN; k0 += 32) {
    v16h a, b;
#pragma unroll
    for (int e = 0; e < 16; ++e) {
      const int ka = k0 + 8 * g + ((e < 8) ? e : (e + 8));   // A-frag K index
      a[e] = (_Float16)x[row * NIN + ka];
      const int kb = k0 + 8 * g + ((e < 8) ? e : (e + 8));   // B-frag K index
      b[e] = (_Float16)W1[(kb + wrow_off) * HID + wcol];
    }
    c = __builtin_amdgcn_wmma_f32_16x16x32_f16(false, a, false, b,
                                               (short)0, c, false, false);
  }
#pragma unroll
  for (int r = 0; r < 8; ++r)
    AB[(tm * 16 + 8 * g + r) * NAB + n] = c[r];
}

// ---------------------------------------------------------------------------
// Kernel 2 (dominant): S[i][h] = (1/1023) * ( sum_p tanh(A[i][h]+B[p][h]+b1[h])
//                                             - tanh(A[i][h]+B[i][h]+b1[h]) )
// 128 threads = one hidden channel each; 4 entity rows per block so each
// B[p][h] load feeds 4 independent tanh ops (good trans-pipe ILP).
// Grid: 1024/4 = 256 blocks x 128 threads. B part streams from L2 (512 KB).
// ---------------------------------------------------------------------------
__global__ __launch_bounds__(128) void k2_tanh_sum(
    const float* __restrict__ AB, const float* __restrict__ b1,
    float* __restrict__ S) {
  const int h  = threadIdx.x;        // 0..127
  const int i0 = blockIdx.x * 4;
  const float bb = b1[h];
  const float a0 = AB[(i0 + 0) * NAB + h] + bb;
  const float a1 = AB[(i0 + 1) * NAB + h] + bb;
  const float a2 = AB[(i0 + 2) * NAB + h] + bb;
  const float a3 = AB[(i0 + 3) * NAB + h] + bb;

  const float* __restrict__ Bq = AB + HID + h;   // B[p][h] at Bq[p*NAB]
  float s0 = 0.f, s1 = 0.f, s2 = 0.f, s3 = 0.f;
#pragma unroll 4
  for (int p = 0; p < NE; ++p) {
    const float bq = Bq[p * NAB];
    s0 += fast_tanh(a0 + bq);
    s1 += fast_tanh(a1 + bq);
    s2 += fast_tanh(a2 + bq);
    s3 += fast_tanh(a3 + bq);
  }
  // remove the p == i diagonal term (reference excludes self-pair)
  s0 -= fast_tanh(a0 + Bq[(i0 + 0) * NAB]);
  s1 -= fast_tanh(a1 + Bq[(i0 + 1) * NAB]);
  s2 -= fast_tanh(a2 + Bq[(i0 + 2) * NAB]);
  s3 -= fast_tanh(a3 + Bq[(i0 + 3) * NAB]);

  const float inv = 1.0f / 1023.0f;
  S[(i0 + 0) * HID + h] = s0 * inv;
  S[(i0 + 1) * HID + h] = s1 * inv;
  S[(i0 + 2) * HID + h] = s2 * inv;
  S[(i0 + 3) * HID + h] = s3 * inv;
}

// ---------------------------------------------------------------------------
// Kernel 3: out[1024 x 64] = S[1024 x 128] @ W2[128 x 64] + b2
// One wave per 16x16 tile, K = 128 -> 4 WMMA steps.
// Grid: 256 tiles / 4 waves = 64 blocks x 128 threads.
// ---------------------------------------------------------------------------
__global__ __launch_bounds__(128) void k3_gemm_sw2(
    const float* __restrict__ S, const float* __restrict__ W2,
    const float* __restrict__ b2, float* __restrict__ out) {
  const int lane = threadIdx.x & 31;
  const int wave = blockIdx.x * (blockDim.x >> 5) + (threadIdx.x >> 5);
  const int tm = wave >> 2;          // 0..63
  const int tn = wave & 3;           // 0..3
  const int g  = lane >> 4;
  const int l  = lane & 15;
  const int row = tm * 16 + l;
  const int n   = tn * 16 + l;       // 0..63

  v8f c = {};
#pragma unroll
  for (int k0 = 0; k0 < HID; k0 += 32) {
    v16h a, b;
#pragma unroll
    for (int e = 0; e < 16; ++e) {
      const int ka = k0 + 8 * g + ((e < 8) ? e : (e + 8));
      a[e] = (_Float16)S[row * HID + ka];
      const int kb = k0 + 8 * g + ((e < 8) ? e : (e + 8));
      b[e] = (_Float16)W2[kb * NOUT + n];
    }
    c = __builtin_amdgcn_wmma_f32_16x16x32_f16(false, a, false, b,
                                               (short)0, c, false, false);
  }
  const float bias = b2[n];
#pragma unroll
  for (int r = 0; r < 8; ++r)
    out[(tm * 16 + 8 * g + r) * NOUT + n] = c[r] + bias;
}

extern "C" void kernel_launch(void* const* d_in, const int* in_sizes, int n_in,
                              void* d_out, int out_size, void* d_ws, size_t ws_size,
                              hipStream_t stream) {
  (void)in_sizes; (void)n_in; (void)out_size; (void)ws_size;
  const float* x  = (const float*)d_in[0];   // [1024, 64]
  const float* W1 = (const float*)d_in[1];   // [128, 128]
  const float* b1 = (const float*)d_in[2];   // [128]
  const float* W2 = (const float*)d_in[3];   // [128, 64]
  const float* b2 = (const float*)d_in[4];   // [64]
  float* out = (float*)d_out;                // [1024, 64]

  float* AB = (float*)d_ws;                  // [1024, 256] = 1 MB
  float* S  = AB + NE * NAB;                 // [1024, 128] = 512 KB

  k1_gemm_xw1<<<256, 128, 0, stream>>>(x, W1, AB);
  k2_tanh_sum<<<256, 128, 0, stream>>>(AB, b1, S);
  k3_gemm_sw2<<<64, 128, 0, stream>>>(S, W2, b2, out);
}
